// NodePairRemovalDecoder_35802847380175
// MI455X (gfx1250) — matrix operand order repaired and checked by reading, not compiled, not run on hardware
//
#include <hip/hip_runtime.h>

#define B_    256
#define N_    1001
#define D_    128
#define H_    4
#define HALF_ 500

typedef __attribute__((ext_vector_type(16))) __bf16 v16bf;
typedef __attribute__((ext_vector_type(8)))  __bf16 v8bf;
typedef __attribute__((ext_vector_type(4)))  __bf16 v4bf;
typedef __attribute__((ext_vector_type(8)))  float  v8f;
typedef __attribute__((ext_vector_type(4)))  float  v4f;
typedef __attribute__((ext_vector_type(4)))  unsigned int u32x4;
typedef __attribute__((ext_vector_type(8)))  int    i32x8;
typedef __attribute__((ext_vector_type(4)))  int    i32x4;

// ---------------------------------------------------------------------------
// Kernel 1: h_hat fp32 -> bf16 copy (halves GEMM + gather read traffic)
// ---------------------------------------------------------------------------
__global__ void cvt_bf16_kernel(const float* __restrict__ x,
                                __bf16* __restrict__ y, int n4) {
  int i = blockIdx.x * blockDim.x + threadIdx.x;
  if (i >= n4) return;
  v4f v = ((const v4f*)x)[i];
  v4bf o;
  o[0] = (__bf16)v.x; o[1] = (__bf16)v.y; o[2] = (__bf16)v.z; o[3] = (__bf16)v.w;
  ((v4bf*)y)[i] = o;
}

// ---------------------------------------------------------------------------
// Kernel 2: Mt[h][j][i] = sum_d WQ[h][i][d] * WK[h][j][d]  (bf16, d-major-K)
// ---------------------------------------------------------------------------
__global__ void build_Mt_kernel(const float* __restrict__ WQ,
                                const float* __restrict__ WK,
                                __bf16* __restrict__ Mt) {
  int t = blockIdx.x * blockDim.x + threadIdx.x;
  if (t >= H_ * D_ * D_) return;
  int h = t / (D_ * D_);
  int r = t % (D_ * D_);
  int j = r / D_;     // output column d of z (row of Mt)
  int i = r % D_;     // contraction index k
  const float* q = WQ + ((long)h * D_ + i) * D_;
  const float* k = WK + ((long)h * D_ + j) * D_;
  float s = 0.f;
#pragma unroll 8
  for (int d = 0; d < D_; ++d) s += q[d] * k[d];
  Mt[t] = (__bf16)s;
}

// ---------------------------------------------------------------------------
// Kernel 3: permutation chains  a1 = s(m), a2 = s^2(m), a3 = s^3(m)
// ---------------------------------------------------------------------------
__global__ void build_idx_kernel(const int* __restrict__ sol,
                                 int* __restrict__ a1, int* __restrict__ a2,
                                 int* __restrict__ a3) {
  int t = blockIdx.x * blockDim.x + threadIdx.x;
  if (t >= B_ * N_) return;
  int b = t / N_, m = t % N_;
  const int* s = sol + (long)b * N_;
  int i1 = s[m];
  int i2 = s[i1];
  int i3 = s[i2];
  a1[t] = i1; a2[t] = i2; a3[t] = i3;
}

// ---------------------------------------------------------------------------
// TDM: stage one 32KB Mt[h] (16384 x bf16, contiguous) into LDS.
// D# built per ISA ch.8: group0 = {count=1 | lds_addr | global_addr | type=2},
// group1 = {data_size=2B, tensor_dim0=16384, tensor_dim1=1, tile_dim0=16384}.
// 6-arg toolchain variant: (u32x4, i32x8, i32x4, i32x4, i32x8, i32 cpol).
// ---------------------------------------------------------------------------
__device__ __forceinline__ void tdm_load_mt(const __bf16* gsrc, unsigned lds_off) {
  unsigned long long ga = (unsigned long long)(size_t)gsrc;
  u32x4 g0;
  g0[0] = 1u;                                              // count=1 (valid)
  g0[1] = lds_off;                                         // lds_addr (bytes)
  g0[2] = (unsigned)(ga & 0xFFFFFFFFu);                    // global_addr[31:0]
  g0[3] = (unsigned)((ga >> 32) & 0x1FFFFFFu) | (2u << 30);// [56:32] | type=2
  i32x8 g1;
  g1[0] = 0x00010000;            // workgroup_mask=0, data_size=1 (2 bytes)
  g1[1] = (int)(16384u << 16);   // tensor_dim0[15:0] in bits[63:48]
  g1[2] = 0x00010000;            // tensor_dim0[31:16]=0 | tensor_dim1[15:0]=1
  g1[3] = (int)(16384u << 16);   // tensor_dim1[31:16]=0 | tile_dim0=16384
  g1[4] = 0;                     // tile_dim1=0 (unused), tile_dim2=0
  g1[5] = 16384;                 // tensor_dim0_stride[31:0]
  g1[6] = 0;                     // stride[47:32]=0, tensor_dim1_stride lo = 0
  g1[7] = 0;
  i32x4 z4 = {0, 0, 0, 0};
  i32x8 z8 = {0, 0, 0, 0, 0, 0, 0, 0};
  __builtin_amdgcn_tensor_load_to_lds(g0, g1, z4, z4, z8, 0);
}

// ---------------------------------------------------------------------------
// Kernel 4: fused  z = h_hat * M[h]  (WMMA bf16)  +  gathered dot epilogue.
//   block = 64 threads = 2 waves; wave owns a 16-row m-tile of one batch.
//   Per head: wave0 TDM-stages Mt[h] (32KB) into LDS, then both waves run
//   8 d-tiles x 4 k-steps of V_WMMA_F32_16X16X32_BF16 with B from LDS.
//   partA[h][b][m]    = z_m . x_{s^2(m)}                 (term2, stored at m)
//   partB[h][b][s(m)] = z_m.x_{s(m)} - z_m.x_{s^3(m)}    (term1-term3, scatter)
// ---------------------------------------------------------------------------
__global__ __launch_bounds__(64)
void fused_z_compat_kernel(const __bf16* __restrict__ hb,
                           const __bf16* __restrict__ Mt,
                           const int* __restrict__ A1,
                           const int* __restrict__ A2,
                           const int* __restrict__ A3,
                           float* __restrict__ partA,
                           float* __restrict__ partB) {
  __shared__ float zsh[2 * H_ * 16 * D_];            // 64 KB z strips
  __shared__ __align__(32) __bf16 mtsh[D_ * D_];     // 32 KB Mt[h] stage

  const int b    = blockIdx.x;          // batch
  const int q    = blockIdx.y;          // tile-pair index 0..31
  const int w    = threadIdx.x >> 5;    // wave in block (0..1)
  const int l    = threadIdx.x & 31;    // lane
  const int tile = q * 2 + w;           // m-tile 0..63 (63 fully masked)
  const int r    = l & 15;              // row within tile / column within d-tile
  const int hi   = l >> 4;              // lane half

  const long m      = (long)tile * 16 + r;
  const long mclamp = (m < N_) ? m : (N_ - 1);
  const __bf16* arow = hb + ((long)b * N_ + mclamp) * D_;

  // ---- A fragments (16x32 bf16): 16 contiguous K at
  //      k0=(lane/16)*16; stay in registers across all heads.
  v16bf aF[4];
#pragma unroll
  for (int kk = 0; kk < 4; ++kk) {
    const int ka = kk * 32 + hi * 16;
    v8bf lo  = *(const v8bf*)(arow + ka);
    v8bf hi8 = *(const v8bf*)(arow + ka + 8);
    v16bf a;
#pragma unroll
    for (int t = 0; t < 8; ++t) { a[t] = lo[t]; a[8 + t] = hi8[t]; }
    aF[kk] = a;
  }

  float* zw = zsh + (long)w * H_ * 16 * D_;
  const unsigned mt_off = (unsigned)(size_t)(void*)mtsh;

  // ---- GEMM: per head, TDM stage of Mt[h] then 8 d-tiles x 4 k WMMAs
#pragma unroll
  for (int h = 0; h < H_; ++h) {
    if (w == 0) {
      tdm_load_mt(Mt + (long)h * D_ * D_, mt_off);
      __builtin_amdgcn_s_wait_tensorcnt(0);
    }
    __syncthreads();                    // Mt[h] visible to both waves
    float* zdst = zw + h * 16 * D_;
#pragma unroll
    for (int dt = 0; dt < 8; ++dt) {
      // B fragment (32x16): lane holds column n=r, 16 contiguous K at
      // (lane/16)*16; Mt is d-major so the K-run is contiguous in LDS.
      const __bf16* brow = mtsh + (dt * 16 + r) * D_;
      v8f c = {};
#pragma unroll
      for (int kk = 0; kk < 4; ++kk) {
        v16bf bF = *(const v16bf*)(brow + kk * 32 + hi * 16);
        c = __builtin_amdgcn_wmma_f32_16x16x32_bf16(
                false, aF[kk], false, bF, (short)0, c, false, false);
      }
      // C layout: VGPR j -> row j + hi*8, col dt*16 + r
#pragma unroll
      for (int j = 0; j < 8; ++j)
        zdst[(j + hi * 8) * D_ + dt * 16 + r] = c[j];
    }
    __syncthreads();                    // protect mtsh before next head's TDM
  }

  // ---- Epilogue: lane pair (l, l+16) shares row r of the tile and splits
  //      the 128-d dot into two 64-d halves, combined with one shfl_xor(16).
  const bool valid = (m < N_);
  const long t0 = (long)b * N_ + (valid ? m : 0);
  const int i1 = A1[t0], i2 = A2[t0], i3 = A3[t0];
  const __bf16* g1 = hb + ((long)b * N_ + i1) * D_ + hi * 64;
  const __bf16* g2 = hb + ((long)b * N_ + i2) * D_ + hi * 64;
  const __bf16* g3 = hb + ((long)b * N_ + i3) * D_ + hi * 64;
  const float*  zr = zw + r * D_ + hi * 64;

  float acc[12];
#pragma unroll
  for (int k = 0; k < 12; ++k) acc[k] = 0.f;

#pragma unroll
  for (int c8 = 0; c8 < 8; ++c8) {
    v8bf G1 = *(const v8bf*)(g1 + c8 * 8);
    v8bf G2 = *(const v8bf*)(g2 + c8 * 8);
    v8bf G3 = *(const v8bf*)(g3 + c8 * 8);
#pragma unroll
    for (int t8 = 0; t8 < 8; ++t8) {
      const float f1 = (float)G1[t8];
      const float f2 = (float)G2[t8];
      const float f3 = (float)G3[t8];
#pragma unroll
      for (int h = 0; h < H_; ++h) {
        const float zv = zr[h * 16 * D_ + c8 * 8 + t8];
        acc[h * 3 + 0] += zv * f1;
        acc[h * 3 + 1] += zv * f2;
        acc[h * 3 + 2] += zv * f3;
      }
    }
  }
#pragma unroll
  for (int k = 0; k < 12; ++k) acc[k] += __shfl_xor(acc[k], 16, 32);

  if (valid && hi == 0) {
#pragma unroll
    for (int h = 0; h < H_; ++h) {
      partA[((long)h * B_ + b) * N_ + m]  = acc[h * 3 + 1];                  // t2
      partB[((long)h * B_ + b) * N_ + i1] = acc[h * 3 + 0] - acc[h * 3 + 2]; // t1-t3
    }
  }
}

// ---------------------------------------------------------------------------
// Kernel 5: feature assembly + 12->32->32->1 MLP (tiny; weights in LDS)
// ---------------------------------------------------------------------------
__global__ void mlp_kernel(const float* __restrict__ pA,
                           const float* __restrict__ pB,
                           const float* __restrict__ sel,
                           const float* __restrict__ w1, const float* __restrict__ b1,
                           const float* __restrict__ w2, const float* __restrict__ b2,
                           const float* __restrict__ w3, const float* __restrict__ b3,
                           float* __restrict__ out) {
  __shared__ float W1[12 * 32], Bi1[32], W2[32 * 32], Bi2[32], W3[32], B3s;
  const int tid = threadIdx.x;
  for (int i = tid; i < 12 * 32; i += blockDim.x) W1[i] = w1[i];
  for (int i = tid; i < 32 * 32; i += blockDim.x) W2[i] = w2[i];
  if (tid < 32) { Bi1[tid] = b1[tid]; Bi2[tid] = b2[tid]; W3[tid] = w3[tid]; }
  if (tid == 0) B3s = b3[0];
  __syncthreads();

  const long g = (long)blockIdx.x * blockDim.x + tid;
  if (g >= (long)B_ * HALF_) return;
  const int b = (int)(g / HALF_);
  const int j = (int)(g % HALF_);

  float feat[12];
#pragma unroll
  for (int c = 0; c < 4; ++c) {
    const long iA = ((long)c * B_ + b) * N_ + (j + 1);    // compat[:, :, 1:501)
    const long iB = ((long)c * B_ + b) * N_ + (j + 501);  // compat[:, :, 501:1001)
    feat[c]     = pA[iA] + pB[iA];
    feat[4 + c] = pA[iB] + pB[iB];
  }
#pragma unroll
  for (int rsel = 0; rsel < 4; ++rsel)
    feat[8 + rsel] = sel[((long)b * 4 + rsel) * HALF_ + j];

  float h1[32];
#pragma unroll 4
  for (int o = 0; o < 32; ++o) {
    float s = Bi1[o];
#pragma unroll
    for (int i = 0; i < 12; ++i) s += feat[i] * W1[i * 32 + o];
    h1[o] = s > 0.f ? s : 0.f;
  }
  float r = B3s;
#pragma unroll 4
  for (int o = 0; o < 32; ++o) {
    float s = Bi2[o];
#pragma unroll 8
    for (int i = 0; i < 32; ++i) s += h1[i] * W2[i * 32 + o];
    s = s > 0.f ? s : 0.f;
    r += s * W3[o];
  }
  out[g] = r;
}

// ---------------------------------------------------------------------------
extern "C" void kernel_launch(void* const* d_in, const int* in_sizes, int n_in,
                              void* d_out, int out_size, void* d_ws, size_t ws_size,
                              hipStream_t stream) {
  const float* h_hat = (const float*)d_in[0];
  const int*   sol   = (const int*)d_in[1];
  const float* sel   = (const float*)d_in[2];
  const float* WQ    = (const float*)d_in[3];
  const float* WK    = (const float*)d_in[4];
  const float* w1    = (const float*)d_in[5];
  const float* b1    = (const float*)d_in[6];
  const float* w2    = (const float*)d_in[7];
  const float* b2    = (const float*)d_in[8];
  const float* w3    = (const float*)d_in[9];
  const float* b3    = (const float*)d_in[10];
  float* out = (float*)d_out;

  // workspace carve-up (~77 MB total, all partitions 32B-aligned)
  char* ws = (char*)d_ws;
  __bf16* hb = (__bf16*)ws;  ws += (size_t)B_ * N_ * D_ * sizeof(__bf16);
  __bf16* Mt = (__bf16*)ws;  ws += (size_t)H_ * D_ * D_ * sizeof(__bf16);
  int* A1 = (int*)ws;        ws += (size_t)B_ * N_ * sizeof(int);
  int* A2 = (int*)ws;        ws += (size_t)B_ * N_ * sizeof(int);
  int* A3 = (int*)ws;        ws += (size_t)B_ * N_ * sizeof(int);
  float* pA = (float*)ws;    ws += (size_t)H_ * B_ * N_ * sizeof(float);
  float* pB = (float*)ws;

  const int n4 = B_ * N_ * D_ / 4;
  cvt_bf16_kernel<<<(n4 + 255) / 256, 256, 0, stream>>>(h_hat, hb, n4);

  build_Mt_kernel<<<(H_ * D_ * D_) / 256, 256, 0, stream>>>(WQ, WK, Mt);

  build_idx_kernel<<<(B_ * N_ + 255) / 256, 256, 0, stream>>>(sol, A1, A2, A3);

  fused_z_compat_kernel<<<dim3(B_, 32), 64, 0, stream>>>(hb, Mt, A1, A2, A3, pA, pB);

  mlp_kernel<<<(B_ * HALF_ + 255) / 256, 256, 0, stream>>>(
      pA, pB, sel, w1, b1, w2, b2, w3, b3, out);
}
